// VAE_22634477650489
// MI455X (gfx1250) — compile-verified, matrix operand order checked by reading, not run on hardware
//
#include <hip/hip_runtime.h>
#include <math.h>

// ---------------- problem constants (from reference) ----------------
#define N_CHARS   50257
#define CHAR_DIM  512
#define HID       1024
#define GATES     (4 * HID)          // 4096
#define SEQ_LEN   2048

// Head rows padded so every block covers 8 waves * 16 rows = 128 rows exactly.
#define HEAD_BLK_ROWS 128
#define N_HEAD_BLOCKS 393                      // 393*128 = 50304
#define NPAD_ROWS     (N_HEAD_BLOCKS * HEAD_BLK_ROWS)

// Zero page length (bf16 elements): must cover kb_max + 24 + 8 <= HID + 32.
#define ZPAD      (HID + 32)

// ---------------- workspace layout (bytes) ----------------
#define OFF_WIH    ((size_t)0)                                    // 4096*512*2  = 4 MiB
#define OFF_WHH    (OFF_WIH   + (size_t)GATES * CHAR_DIM * 2)     // 4096*1024*2 = 8 MiB
#define OFF_WHEAD  (OFF_WHH   + (size_t)GATES * HID * 2)          // 50304*1024*2 ~= 103 MB
#define OFF_H      (OFF_WHEAD + (size_t)NPAD_ROWS * HID * 2)
#define OFF_C      (OFF_H     + (size_t)HID * 4)
#define OFF_X      (OFF_C     + (size_t)HID * 4)
#define OFF_PVAL   (OFF_X     + (size_t)CHAR_DIM * 4)
#define OFF_PIDX   (OFF_PVAL  + (size_t)512 * 4)

typedef __attribute__((ext_vector_type(16))) __bf16 v16bf;
typedef __attribute__((ext_vector_type(8)))  __bf16 v8bf;
typedef __attribute__((ext_vector_type(8)))  float  v8f;

// ---------------- helpers ----------------
__device__ __forceinline__ unsigned short f2bf(float f) {
    unsigned int u = __builtin_bit_cast(unsigned int, f);
    u += 0x7FFFu + ((u >> 16) & 1u);          // round-to-nearest-even
    return (unsigned short)(u >> 16);
}

union U128 { uint4 u; v8bf b; };

// Concatenate two 16B chunks (8 bf16 each) into a v16bf fragment.
__device__ __forceinline__ v16bf frag16(const unsigned short* p, int kb) {
    U128 lo, hi;
    lo.u = *reinterpret_cast<const uint4*>(p + kb);        // 8 bf16 = 16B
    hi.u = *reinterpret_cast<const uint4*>(p + kb + 16);   // 8 bf16 = 16B
    v16bf r;
#pragma unroll
    for (int i = 0; i < 8; ++i) { r[i] = lo.b[i]; r[8 + i] = hi.b[i]; }
    return r;
}

__device__ __forceinline__ float sigmoidf_(float x) { return 1.0f / (1.0f + expf(-x)); }

// ---------------- setup kernels ----------------
__global__ void cvt_f32_bf16_kernel(const float* __restrict__ src,
                                    unsigned short* __restrict__ dst,
                                    long n, long nvalid) {
    long i = (long)blockIdx.x * blockDim.x + threadIdx.x;
    long stride = (long)gridDim.x * blockDim.x;
    for (; i < n; i += stride)
        dst[i] = (i < nvalid) ? f2bf(src[i]) : (unsigned short)0;
}

__global__ void init_state_kernel(const float* __restrict__ inputs,
                                  const float* __restrict__ embed,
                                  const int* __restrict__ start,
                                  float* __restrict__ h, float* __restrict__ c,
                                  float* __restrict__ x) {
    const int i = threadIdx.x;                       // blockDim = 1024
    h[i] = inputs[i];
    c[i] = 0.0f;
    if (i < CHAR_DIM) x[i] = embed[(size_t)(*start) * CHAR_DIM + i];
}

// ---------------- fused LSTM cell: gates (WMMA) + activations + state ----------------
// grid = HID/16 = 64 blocks, block = 128 threads (4 waves; wave g handles gate g).
__global__ void lstm_cell_kernel(const unsigned short* __restrict__ Wih,
                                 const unsigned short* __restrict__ Whh,
                                 const float* __restrict__ b_ih,
                                 const float* __restrict__ b_hh,
                                 const float* __restrict__ x,
                                 float* __restrict__ h,
                                 float* __restrict__ c) {
    __shared__ __align__(16) unsigned short vec[CHAR_DIM + HID];   // x ++ h as bf16
    __shared__ __align__(16) unsigned short zbuf[ZPAD];            // zero page for B
    __shared__ float gbuf[4][16];

    const int tid  = threadIdx.x;
    const int gate = tid >> 5;
    const int lane = tid & 31;
    const int blk  = blockIdx.x;

    for (int i = tid; i < CHAR_DIM; i += 128) vec[i] = f2bf(x[i]);
    for (int i = tid; i < HID;      i += 128) vec[CHAR_DIM + i] = f2bf(h[i]);
    for (int i = tid; i < ZPAD;     i += 128) zbuf[i] = 0;
    __syncthreads();

    const int m      = lane & 15;
    const int half   = (lane & 16) ? 8 : 0;
    const bool contrib = ((lane & 15) == 0);         // lanes 0 and 16 carry column N=0
    const int grow   = gate * HID + blk * 16 + m;    // row in [0,4096)

    // B base pointers: contributing lanes walk the real vector, others walk zeros.
    const unsigned short* vx = contrib ? (vec + half)            : zbuf;
    const unsigned short* vh = contrib ? (vec + CHAR_DIM + half) : zbuf;

    v8f acc = {};
    const unsigned short* rih = Wih + (size_t)grow * CHAR_DIM + half;
    for (int kb = 0; kb < CHAR_DIM; kb += 32) {
        v16bf a = frag16(rih, kb);                   // A: 16x32 weight tile slice
        v16bf b = frag16(vx, kb);                    // B: vector in column 0, zeros elsewhere
        acc = __builtin_amdgcn_wmma_f32_16x16x32_bf16(false, a, false, b,
                                                      (short)0, acc, false, false);
    }
    const unsigned short* rhh = Whh + (size_t)grow * HID + half;
    for (int kb = 0; kb < HID; kb += 32) {
        v16bf a = frag16(rhh, kb);
        v16bf b = frag16(vh, kb);
        acc = __builtin_amdgcn_wmma_f32_16x16x32_bf16(false, a, false, b,
                                                      (short)0, acc, false, false);
    }

    // Column 0 of the 16x16 f32 accumulator: lane 0 holds M=0..7, lane 16 holds M=8..15.
    if (contrib) {
        const int rbase = half;                      // 0 for lane 0, 8 for lane 16
#pragma unroll
        for (int j = 0; j < 8; ++j) {
            const int gr = gate * HID + blk * 16 + rbase + j;
            gbuf[gate][rbase + j] = acc[j] + b_ih[gr] + b_hh[gr];
        }
    }
    __syncthreads();

    if (tid < 16) {
        const int r  = blk * 16 + tid;
        const float gi = sigmoidf_(gbuf[0][tid]);
        const float gf = sigmoidf_(gbuf[1][tid]);
        const float gg = tanhf(gbuf[2][tid]);
        const float go = sigmoidf_(gbuf[3][tid]);
        const float cn = gf * c[r] + gi * gg;
        c[r] = cn;
        h[r] = go * tanhf(cn);
    }
}

// ---------------- head GEMV (WMMA) + logits + block argmax ----------------
// grid = 393 blocks, block = 256 threads (8 waves, 16 rows each => 128 rows/block).
__global__ void head_kernel(const unsigned short* __restrict__ Whead,
                            const float* __restrict__ b_head,
                            const float* __restrict__ h,
                            float* __restrict__ logits,
                            float* __restrict__ pval,
                            int* __restrict__ pidx) {
    __shared__ __align__(16) unsigned short hv[HID];
    __shared__ __align__(16) unsigned short zbuf[ZPAD];
    __shared__ float sval[16];
    __shared__ int   sidx[16];

    const int tid  = threadIdx.x;
    const int wave = tid >> 5;
    const int lane = tid & 31;

    for (int i = tid; i < HID;  i += 256) hv[i]   = f2bf(h[i]);
    for (int i = tid; i < ZPAD; i += 256) zbuf[i] = 0;
    __syncthreads();

    const int m       = lane & 15;
    const int half    = (lane & 16) ? 8 : 0;
    const bool contrib = ((lane & 15) == 0);
    const int rowTile = blockIdx.x * HEAD_BLK_ROWS + wave * 16;
    const unsigned short* rp = Whead + (size_t)(rowTile + m) * HID + half;
    const unsigned short* vh = contrib ? (hv + half) : zbuf;

    v8f acc = {};
    for (int kb = 0; kb < HID; kb += 32) {
        v16bf a = frag16(rp, kb);
        v16bf b = frag16(vh, kb);
        acc = __builtin_amdgcn_wmma_f32_16x16x32_bf16(false, a, false, b,
                                                      (short)0, acc, false, false);
    }

    if (contrib) {
        const int rbase = half;                     // 0 for lane 0, 8 for lane 16
        const int r0    = rowTile + rbase;          // 8 contiguous rows start here
        float best = -3.4e38f;
        int   bidx = 0;
        float vv[8];
        if (r0 + 7 < N_CHARS) {
            // fast path: vector bias load + two b128 logit stores
            const float4 b0 = *reinterpret_cast<const float4*>(b_head + r0);
            const float4 b1 = *reinterpret_cast<const float4*>(b_head + r0 + 4);
            vv[0] = acc[0] + b0.x; vv[1] = acc[1] + b0.y;
            vv[2] = acc[2] + b0.z; vv[3] = acc[3] + b0.w;
            vv[4] = acc[4] + b1.x; vv[5] = acc[5] + b1.y;
            vv[6] = acc[6] + b1.z; vv[7] = acc[7] + b1.w;
            float4 o0 = { vv[0], vv[1], vv[2], vv[3] };
            float4 o1 = { vv[4], vv[5], vv[6], vv[7] };
            *reinterpret_cast<float4*>(logits + r0)     = o0;
            *reinterpret_cast<float4*>(logits + r0 + 4) = o1;
#pragma unroll
            for (int j = 0; j < 8; ++j)
                if (vv[j] > best) { best = vv[j]; bidx = r0 + j; }  // ascending -> first-max
        } else {
            // tail tile (only the very last block): scalar guarded path
#pragma unroll
            for (int j = 0; j < 8; ++j) {
                const int r = r0 + j;
                if (r < N_CHARS) {
                    const float v = acc[j] + b_head[r];
                    logits[r] = v;
                    if (v > best) { best = v; bidx = r; }
                }
            }
        }
        const int slot = wave * 2 + (lane >> 4);        // ascending row order
        sval[slot] = best;
        sidx[slot] = bidx;
    }
    __syncthreads();

    if (tid == 0) {
        float bv = -3.4e38f; int bi = 0;
#pragma unroll
        for (int k = 0; k < 16; ++k)
            if (sval[k] > bv) { bv = sval[k]; bi = sidx[k]; }
        pval[blockIdx.x] = bv;
        pidx[blockIdx.x] = bi;
    }
}

// ---------------- final argmax + embedding gather ----------------
__global__ void argmax_gather_kernel(const float* __restrict__ pval,
                                     const int* __restrict__ pidx,
                                     const float* __restrict__ embed,
                                     float* __restrict__ x,
                                     float* __restrict__ char_out) {
    __shared__ float sv[512];
    __shared__ int   si[512];
    __shared__ int   bestIdx;
    const int tid = threadIdx.x;                     // blockDim = 512

    sv[tid] = (tid < N_HEAD_BLOCKS) ? pval[tid] : -3.4e38f;
    si[tid] = (tid < N_HEAD_BLOCKS) ? pidx[tid] : 0;
    __syncthreads();

    if (tid == 0) {
        float bv = -3.4e38f; int bi = 0;
        for (int k = 0; k < N_HEAD_BLOCKS; ++k)      // ascending -> first-max kept
            if (sv[k] > bv) { bv = sv[k]; bi = si[k]; }
        bestIdx = bi;
        *char_out = (float)bi;
    }
    __syncthreads();

    if (tid < CHAR_DIM) x[tid] = embed[(size_t)bestIdx * CHAR_DIM + tid];
}

// ---------------- launcher ----------------
extern "C" void kernel_launch(void* const* d_in, const int* in_sizes, int n_in,
                              void* d_out, int out_size, void* d_ws, size_t ws_size,
                              hipStream_t stream) {
    const float* inputs = (const float*)d_in[0];
    const float* embed  = (const float*)d_in[1];
    const float* W_ih   = (const float*)d_in[2];
    const float* W_hh   = (const float*)d_in[3];
    const float* b_ih   = (const float*)d_in[4];
    const float* b_hh   = (const float*)d_in[5];
    const float* W_head = (const float*)d_in[6];
    const float* b_head = (const float*)d_in[7];
    const int*   start  = (const int*)d_in[9];

    char* ws = (char*)d_ws;
    unsigned short* wih_bf   = (unsigned short*)(ws + OFF_WIH);
    unsigned short* whh_bf   = (unsigned short*)(ws + OFF_WHH);
    unsigned short* whead_bf = (unsigned short*)(ws + OFF_WHEAD);
    float* h    = (float*)(ws + OFF_H);
    float* c    = (float*)(ws + OFF_C);
    float* x    = (float*)(ws + OFF_X);
    float* pval = (float*)(ws + OFF_PVAL);
    int*   pidx = (int*)(ws + OFF_PIDX);

    float* chars_out  = (float*)d_out;               // 2048 chars
    float* logits_out = (float*)d_out + SEQ_LEN;     // 2048 x 50257 logits

    // One-time bf16 weight packing (deterministic; makes W_head L2-resident: ~116 MB < 192 MB L2).
    {
        const long n1 = (long)GATES * CHAR_DIM;
        cvt_f32_bf16_kernel<<<1024, 256, 0, stream>>>(W_ih, wih_bf, n1, n1);
        const long n2 = (long)GATES * HID;
        cvt_f32_bf16_kernel<<<1024, 256, 0, stream>>>(W_hh, whh_bf, n2, n2);
        const long n3 = (long)NPAD_ROWS * HID;
        const long v3 = (long)N_CHARS * HID;
        cvt_f32_bf16_kernel<<<4096, 256, 0, stream>>>(W_head, whead_bf, n3, v3);
    }
    init_state_kernel<<<1, HID, 0, stream>>>(inputs, embed, start, h, c, x);

    for (int t = 0; t < SEQ_LEN; ++t) {
        lstm_cell_kernel<<<HID / 16, 128, 0, stream>>>(wih_bf, whh_bf, b_ih, b_hh, x, h, c);
        head_kernel<<<N_HEAD_BLOCKS, 256, 0, stream>>>(whead_bf, b_head, h,
                                                       logits_out + (size_t)t * N_CHARS,
                                                       pval, pidx);
        argmax_gather_kernel<<<1, 512, 0, stream>>>(pval, pidx, embed, x, chars_out + t);
    }
}
